// Attention2D_40853728919700
// MI455X (gfx1250) — compile-verified, matrix operand order checked by reading, not run on hardware
//
#include <hip/hip_runtime.h>

// Geometry (fixed by the reference)
#define Csz   384
#define NHEAD 12
#define HD    32
#define NTOK  64      // tokens per window (8x8)
#define NWIN  64      // windows per image
#define SHFT  4       // cyclic shift
#define LDX   (Csz + 8)   // padded LDS row (in bf16 elems, multiple of 8)

typedef __attribute__((ext_vector_type(16))) __bf16 v16bf;
typedef __attribute__((ext_vector_type(8)))  __bf16 v8bf;
typedef __attribute__((ext_vector_type(4)))  __bf16 v4bf;
typedef __attribute__((ext_vector_type(8)))  float  v8f;

// Load one 16x32 bf16 WMMA operand fragment from a row-major LDS tile.
// CDNA5 16-bit A-layout: lanes 0-15 hold rows M=0..15 with K in
// [k0,k0+8) and [k0+16,k0+24); lanes 16-31 hold the same rows with k0+=8.
// Works for B as well when the tile is stored N-major (i.e. B transposed).
__device__ __forceinline__ v16bf ldfrag(const __bf16* tile, int row0, int kbase,
                                        int ld, int lane) {
  int row = row0 + (lane & 15);
  int kb  = kbase + ((lane >> 4) << 3);
  const __bf16* p = tile + row * ld + kb;
  v8bf lo = *(const v8bf*)(p);
  v8bf hi = *(const v8bf*)(p + 16);
  v16bf r;
#pragma unroll
  for (int i = 0; i < 8; ++i) { r[i] = lo[i]; r[i + 8] = hi[i]; }
  return r;
}

__global__ __launch_bounds__(256)
void swin_window_attn_fused(const float* __restrict__ x,
                            const float* __restrict__ w_qkv,
                            const float* __restrict__ w_proj,
                            const float* __restrict__ b_proj,
                            const float* __restrict__ emb,
                            const int*   __restrict__ bidx,
                            const float* __restrict__ smask,
                            float* __restrict__ out) {
  __shared__ __align__(16) __bf16 lds_x [NTOK * LDX];   // window input, bf16
  __shared__ __align__(16) __bf16 lds_wt[96   * LDX];   // transposed weight slab
  __shared__ __align__(16) __bf16 lds_q [NTOK * HD];    // q (scaled), row-major
  __shared__ __align__(16) __bf16 lds_k [NTOK * HD];    // k, row-major (=B^T for qk^T)
  __shared__ __align__(16) __bf16 lds_vt[HD   * NTOK];  // v transposed (=B^T for a*v)
  __shared__ __align__(16) float  lds_s [NTOK * NTOK];  // attention logits
  __shared__ __align__(16) __bf16 lds_a [NTOK * NTOK];  // softmax probs, bf16
  __shared__ __align__(16) __bf16 lds_o [NTOK * LDX];   // pre-proj output

  const int tid  = threadIdx.x;
  const int lane = tid & 31;
  const int wv   = tid >> 5;            // 8 waves
  const int win  = blockIdx.x;          // B*NW blocks
  const int bimg = win >> 6;            // image
  const int w    = win & 63;            // window within image
  const int wy = w >> 3, wx = w & 7;
  const int nloc  = lane & 15;
  const int mhalf = (lane >> 4) << 3;   // C/D fragment: M = mi*16 + mhalf + r
  const float scale = 0.0510310363f;    // 384^-0.5 (ref scales by embed_dim)

  // ---- gather cyclically-shifted window into LDS, f32 -> bf16 (b128 loads) ----
  for (int i = tid; i < NTOK * (Csz / 4); i += 256) {
    int n = i / (Csz / 4), c = (i - n * (Csz / 4)) << 2;
    int hs = ((wy << 3) + (n >> 3) + SHFT) & 63;
    int ws = ((wx << 3) + (n & 7)  + SHFT) & 63;
    float4 v = *(const float4*)&x[((size_t)bimg * 4096 + hs * 64 + ws) * Csz + c];
    v4bf p = { (__bf16)v.x, (__bf16)v.y, (__bf16)v.z, (__bf16)v.w };
    *(v4bf*)&lds_x[n * LDX + c] = p;
  }

  for (int h = 0; h < NHEAD; ++h) {
    __syncthreads();
    // ---- stage per-head qkv weight slice, N-major: lds_wt[o][k] ----
    // b128 global loads (4 consecutive out-cols share a qkv chunk); LDS side
    // is the transpose scatter (4x b16), which is the cheap side to pay.
    for (int i = tid; i < 96 * Csz / 4; i += 256) {
      int kc = i / 24, o = (i - kc * 24) << 2;
      int col = (o >> 5) * Csz + h * HD + (o & 31);
      float4 v = *(const float4*)&w_qkv[(size_t)kc * (3 * Csz) + col];
      lds_wt[(o + 0) * LDX + kc] = (__bf16)v.x;
      lds_wt[(o + 1) * LDX + kc] = (__bf16)v.y;
      lds_wt[(o + 2) * LDX + kc] = (__bf16)v.z;
      lds_wt[(o + 3) * LDX + kc] = (__bf16)v.w;
    }
    __syncthreads();

    // ---- QKV GEMM: (64x384)*(384x96); 24 tiles, 3 per wave ----
#pragma unroll
    for (int t3 = 0; t3 < 3; ++t3) {
      int t = wv + t3 * 8;
      int mi = t & 3, ni = t >> 2;                 // ni: 0,1=Q 2,3=K 4,5=V
      v8f acc = {};
      for (int kk = 0; kk < Csz; kk += 32) {
        v16bf af = ldfrag(lds_x,  mi * 16, kk, LDX, lane);
        v16bf bf = ldfrag(lds_wt, ni * 16, kk, LDX, lane);
        acc = __builtin_amdgcn_wmma_f32_16x16x32_bf16(false, af, false, bf,
                                                      (short)0, acc, false, false);
      }
      int o0 = ni * 16 + nloc;
      int mb = mi * 16 + mhalf;
      if (ni < 2) {
#pragma unroll
        for (int r = 0; r < 8; ++r)
          lds_q[(mb + r) * HD + o0] = (__bf16)(acc[r] * scale);
      } else if (ni < 4) {
#pragma unroll
        for (int r = 0; r < 8; ++r)
          lds_k[(mb + r) * HD + (o0 - 32)] = (__bf16)acc[r];
      } else {
#pragma unroll
        for (int r = 0; r < 8; ++r)
          lds_vt[(o0 - 64) * NTOK + (mb + r)] = (__bf16)acc[r];
      }
    }
    __syncthreads();

    // ---- s = q * k^T : HD=32 => single K-step; 16 tiles, 2 per wave ----
#pragma unroll
    for (int t2 = 0; t2 < 2; ++t2) {
      int t = wv + t2 * 8;
      int mi = t & 3, ni = t >> 2;
      v16bf qf = ldfrag(lds_q, mi * 16, 0, HD, lane);
      v16bf kf = ldfrag(lds_k, ni * 16, 0, HD, lane);
      v8f z = {};
      v8f sacc = __builtin_amdgcn_wmma_f32_16x16x32_bf16(false, qf, false, kf,
                                                         (short)0, z, false, false);
      int mb = mi * 16 + mhalf;
#pragma unroll
      for (int r = 0; r < 8; ++r)
        lds_s[(mb + r) * NTOK + ni * 16 + nloc] = sacc[r];
    }
    __syncthreads();

    // ---- softmax row-wise, with rel-pos bias + shift mask ----
    if (tid < NTOK) {
      const int n = tid;
      const float* mrow = smask + (size_t)w * (NTOK * NTOK) + n * NTOK;
      const int* ih = bidx + n * NTOK;
      const int* iw = bidx + NTOK * NTOK + n * NTOK;
      float mx = -3.0e38f;
      for (int m = 0; m < NTOK; ++m) {
        float bias = emb[ih[m] * NHEAD + h] + emb[iw[m] * NHEAD + h];
        float v = lds_s[n * NTOK + m] + bias + mrow[m];
        lds_s[n * NTOK + m] = v;
        mx = fmaxf(mx, v);
      }
      float sum = 0.f;
      for (int m = 0; m < NTOK; ++m) {
        float e = __expf(lds_s[n * NTOK + m] - mx);
        lds_s[n * NTOK + m] = e;
        sum += e;
      }
      float inv = 1.0f / sum;
      for (int m = 0; m < NTOK; ++m)
        lds_a[n * NTOK + m] = (__bf16)(lds_s[n * NTOK + m] * inv);
    }
    __syncthreads();

    // ---- o = a * v : (64x64)*(64x32); 8 tiles, 1 per wave ----
    {
      int mi = wv & 3, ni = wv >> 2;
      v8f oacc = {};
#pragma unroll
      for (int kk = 0; kk < NTOK; kk += 32) {
        v16bf afr = ldfrag(lds_a,  mi * 16, kk, NTOK, lane);
        v16bf vfr = ldfrag(lds_vt, ni * 16, kk, NTOK, lane);
        oacc = __builtin_amdgcn_wmma_f32_16x16x32_bf16(false, afr, false, vfr,
                                                       (short)0, oacc, false, false);
      }
      int mb = mi * 16 + mhalf;
#pragma unroll
      for (int r = 0; r < 8; ++r)
        lds_o[(mb + r) * LDX + h * HD + ni * 16 + nloc] = (__bf16)oacc[r];
    }
  } // heads

  // ---- proj: (64x384)*(384x384)+bias, streamed in 4 col-blocks of 96;
  //      reverse cyclic shift folded into the scatter ----
  for (int cb = 0; cb < 4; ++cb) {
    __syncthreads();
    for (int i = tid; i < 96 * Csz / 4; i += 256) {
      int kc = i / 24, o = (i - kc * 24) << 2;
      float4 v = *(const float4*)&w_proj[(size_t)kc * Csz + cb * 96 + o];
      lds_wt[(o + 0) * LDX + kc] = (__bf16)v.x;
      lds_wt[(o + 1) * LDX + kc] = (__bf16)v.y;
      lds_wt[(o + 2) * LDX + kc] = (__bf16)v.z;
      lds_wt[(o + 3) * LDX + kc] = (__bf16)v.w;
    }
    __syncthreads();
#pragma unroll
    for (int t3 = 0; t3 < 3; ++t3) {
      int t = wv + t3 * 8;
      int mi = t & 3, ni = t >> 2;
      v8f acc = {};
      for (int kk = 0; kk < Csz; kk += 32) {
        v16bf afr = ldfrag(lds_o,  mi * 16, kk, LDX, lane);
        v16bf bfr = ldfrag(lds_wt, ni * 16, kk, LDX, lane);
        acc = __builtin_amdgcn_wmma_f32_16x16x32_bf16(false, afr, false, bfr,
                                                      (short)0, acc, false, false);
      }
      int c = cb * 96 + ni * 16 + nloc;
      float bias = b_proj[c];
      int mb = mi * 16 + mhalf;
#pragma unroll
      for (int r = 0; r < 8; ++r) {
        int n = mb + r;
        int hs = ((wy << 3) + (n >> 3) + SHFT) & 63;
        int ws = ((wx << 3) + (n & 7)  + SHFT) & 63;
        out[((size_t)bimg * 4096 + hs * 64 + ws) * Csz + c] = acc[r] + bias;
      }
    }
  }
}

extern "C" void kernel_launch(void* const* d_in, const int* in_sizes, int n_in,
                              void* d_out, int out_size, void* d_ws, size_t ws_size,
                              hipStream_t stream) {
  const float* x      = (const float*)d_in[0];
  const float* w_qkv  = (const float*)d_in[1];
  const float* w_proj = (const float*)d_in[2];
  const float* b_proj = (const float*)d_in[3];
  const float* emb    = (const float*)d_in[4];
  const int*   bidx   = (const int*)d_in[5];
  const float* smask  = (const float*)d_in[6];
  int B = in_sizes[0] / (4096 * Csz);       // 32
  swin_window_attn_fused<<<dim3(B * NWIN), 256, 0, stream>>>(
      x, w_qkv, w_proj, b_proj, emb, bidx, smask, (float*)d_out);
}
